// DecoderLatent_1108101563174
// MI455X (gfx1250) — matrix operand order repaired and checked by reading, not compiled, not run on hardware
//
#include <hip/hip_runtime.h>
#include <hip/hip_bf16.h>

#define L_    2
#define H_    1024
#define Z_    256
#define B_    128
#define SRC_  64
#define KMAX_ 48
#define IN_   1281      // Z + 1 + H
#define KP0_  2336      // pad(IN + H = 2305) -> 73*32
#define KP1_  2048      // H + H
#define G4H_  4096      // 4*H

typedef __attribute__((ext_vector_type(16))) __bf16 v16bf;
typedef __attribute__((ext_vector_type(8)))  __bf16 v8bf;
typedef __attribute__((ext_vector_type(8)))  float  v8f;

#define CEILDIV(a,b) (((a)+(b)-1)/(b))

// ---------------------------------------------------------------------------
// WMMA fragment loaders (CDNA5 16x16x32 bf16 layouts, ISA 7.12.2)
// A (16x32, MxK): lane holds row M=lane&15; K-chunks [k0..k0+7],[k0+16..k0+23],
//                 k0 = (lane>>4)*8.
// B (32x16, KxN): lane holds col N=lane&15; K-chunks [k0..k0+7],[k0+16..k0+23],
//                 k0 = (lane>>4)*8.
// ---------------------------------------------------------------------------
__device__ __forceinline__ v16bf load_a_frag(const __bf16* __restrict__ X, int ldx,
                                             int mBase, int k, int lane) {
    const int m = mBase + (lane & 15);
    const __bf16* p = X + (size_t)m * ldx + k + ((lane >> 4) << 3);
    v8bf c0 = *reinterpret_cast<const v8bf*>(p);
    v8bf c1 = *reinterpret_cast<const v8bf*>(p + 16);
    v16bf a;
#pragma unroll
    for (int i = 0; i < 8; ++i) { a[i] = c0[i]; a[i + 8] = c1[i]; }
    return a;
}

__device__ __forceinline__ v16bf load_b_frag(const __bf16* __restrict__ W, int ldw,
                                             int nBase, int k, int lane) {
    const int n = nBase + (lane & 15);
    const __bf16* p = W + (size_t)n * ldw + k + ((lane >> 4) << 3);
    v8bf c0 = *reinterpret_cast<const v8bf*>(p);
    v8bf c1 = *reinterpret_cast<const v8bf*>(p + 16);
    v16bf b;
#pragma unroll
    for (int i = 0; i < 8; ++i) { b[i] = c0[i]; b[i + 8] = c1[i]; }
    return b;
}

__device__ __forceinline__ float sigm(float x) { return 1.0f / (1.0f + expf(-x)); }

// ---------------------------------------------------------------------------
// Generic WMMA GEMM: Y[M=128,N] = act(X[128,Kp]_bf16 @ W[N,Kp]^T_bf16 + bias)
// One wave (32 threads) per 16x16 output tile. ACT: 0=none, 1=tanh.
// ---------------------------------------------------------------------------
template <int ACT>
__global__ void wmma_gemm_kernel(const __bf16* __restrict__ X, int ldx,
                                 const __bf16* __restrict__ W, int Kp,
                                 const float* __restrict__ bias,
                                 float* __restrict__ Y, int ldy) {
    const int lane  = threadIdx.x;
    const int nBase = blockIdx.x * 16;
    const int mBase = blockIdx.y * 16;
    v8f acc = {};
    for (int k = 0; k < Kp; k += 32) {
        v16bf a = load_a_frag(X, ldx, mBase, k, lane);
        v16bf b = load_b_frag(W, Kp, nBase, k, lane);
        acc = __builtin_amdgcn_wmma_f32_16x16x32_bf16(false, a, false, b,
                                                      (short)0, acc, false, false);
    }
    const int n      = nBase + (lane & 15);
    const int rowOff = (lane >> 4) * 8;
    const float bv   = bias ? bias[n] : 0.0f;
#pragma unroll
    for (int r = 0; r < 8; ++r) {
        const int m = mBase + rowOff + r;
        float v = acc[r] + bv;
        if (ACT == 1) v = tanhf(v);
        Y[(size_t)m * ldy + n] = v;
    }
}

// ---------------------------------------------------------------------------
// Fused LSTM gate GEMM + cell. W is [4H, Kp] (gates i,f,g,o stacked on rows).
// Each wave computes the same 16x16 (m,n) tile for all four gates: the A
// fragment is loaded once per K-slice and feeds 4 back-to-back WMMAs.
// State update is frozen per batch row when step >= k[m]; raw h goes to h_raw.
// ---------------------------------------------------------------------------
__global__ void wmma_lstm_kernel(const __bf16* __restrict__ X, int Kp,
                                 const __bf16* __restrict__ W,
                                 const float* __restrict__ bias,
                                 const int* __restrict__ kv, int step,
                                 float* __restrict__ h_state,
                                 float* __restrict__ c_state,
                                 float* __restrict__ h_raw) {
    const int lane  = threadIdx.x;
    const int nBase = blockIdx.x * 16;
    const int mBase = blockIdx.y * 16;
    v8f acc[4] = {};
    for (int k = 0; k < Kp; k += 32) {
        v16bf a = load_a_frag(X, Kp, mBase, k, lane);
#pragma unroll
        for (int g = 0; g < 4; ++g) {
            v16bf b = load_b_frag(W + (size_t)g * H_ * Kp, Kp, nBase, k, lane);
            acc[g] = __builtin_amdgcn_wmma_f32_16x16x32_bf16(false, a, false, b,
                                                             (short)0, acc[g], false, false);
        }
    }
    const int n      = nBase + (lane & 15);
    const int rowOff = (lane >> 4) * 8;
#pragma unroll
    for (int r = 0; r < 8; ++r) {
        const int m = mBase + rowOff + r;
        const float gi = acc[0][r] + bias[n];
        const float gf = acc[1][r] + bias[H_ + n];
        const float gg = acc[2][r] + bias[2 * H_ + n];
        const float go = acc[3][r] + bias[3 * H_ + n];
        const size_t idx = (size_t)m * H_ + n;
        const float cp = c_state[idx];
        const float cn = sigm(gf) * cp + sigm(gi) * tanhf(gg);
        const float hn = sigm(go) * tanhf(cn);
        h_raw[idx] = hn;
        if (step < kv[m]) { h_state[idx] = hn; c_state[idx] = cn; }
    }
}

// ---------------------------------------------------------------------------
// Per-step packing / elementwise kernels
// ---------------------------------------------------------------------------
__global__ void pack_x0_kernel(const float* __restrict__ zi, const int* __restrict__ kv,
                               const float* __restrict__ attn, const float* __restrict__ h0s,
                               __bf16* __restrict__ X0) {
    const int i = blockIdx.x * blockDim.x + threadIdx.x;
    if (i >= B_ * KP0_) return;
    const int b = i / KP0_, kk = i % KP0_;
    float v;
    if (kk < Z_)              v = zi[b * Z_ + kk];
    else if (kk == Z_)        v = (float)kv[b];
    else if (kk < IN_)        v = attn[b * H_ + (kk - Z_ - 1)];
    else if (kk < IN_ + H_)   v = h0s[(size_t)b * H_ + (kk - IN_)];
    else                      v = 0.0f;
    X0[i] = (__bf16)v;
}

__global__ void pack_x1_kernel(const float* __restrict__ h0r, const float* __restrict__ h1s,
                               __bf16* __restrict__ X1) {
    const int i = blockIdx.x * blockDim.x + threadIdx.x;
    if (i >= B_ * KP1_) return;
    const int b = i / KP1_, kk = i % KP1_;
    const float v = (kk < H_) ? h0r[(size_t)b * H_ + kk]
                              : h1s[(size_t)b * H_ + (kk - H_)];
    X1[i] = (__bf16)v;
}

// write f32 [B,H] source into the high half (cols H..2H) of Xcat [B,2H] bf16
__global__ void pack_xcat_hi_kernel(const float* __restrict__ src, __bf16* __restrict__ xc) {
    const int i = blockIdx.x * blockDim.x + threadIdx.x;
    if (i >= B_ * H_) return;
    const int b = i / H_, h = i % H_;
    xc[(size_t)b * (2 * H_) + H_ + h] = (__bf16)src[i];
}

// scores + softmax + weighted sum; weighted goes to low half of Xcat as bf16.
__global__ void attn_scores_kernel(const float* __restrict__ ctx,
                                   const float* __restrict__ qin,
                                   __bf16* __restrict__ xc) {
    __shared__ float al[SRC_];
    const int b = blockIdx.x, t = threadIdx.x;
    const float* cb = ctx + (size_t)b * SRC_ * H_;
    const float* qb = qin + (size_t)b * H_;
    const int s = t >> 2, sub = t & 3;
    float p = 0.0f;
    for (int h = sub; h < H_; h += 4) p += cb[(size_t)s * H_ + h] * qb[h];
    p += __shfl_xor(p, 1, 4);
    p += __shfl_xor(p, 2, 4);
    if (sub == 0) al[s] = p;
    __syncthreads();
    if (t == 0) {
        float mx = al[0];
        for (int i = 1; i < SRC_; ++i) mx = fmaxf(mx, al[i]);
        float sm = 0.0f;
        for (int i = 0; i < SRC_; ++i) { float e = expf(al[i] - mx); al[i] = e; sm += e; }
        const float inv = 1.0f / sm;
        for (int i = 0; i < SRC_; ++i) al[i] *= inv;
    }
    __syncthreads();
    for (int h = t; h < H_; h += 256) {
        float w = 0.0f;
        for (int i = 0; i < SRC_; ++i) w += al[i] * cb[(size_t)i * H_ + h];
        xc[(size_t)b * (2 * H_) + h] = (__bf16)w;
    }
}

// z_next = mu + exp(sigma)*eps; write masked z/mu/sigma outputs, carry raw z.
__global__ void zstep_kernel(const float* __restrict__ musig, const float* __restrict__ eps_i,
                             const int* __restrict__ kv, int step,
                             float* __restrict__ zi,
                             float* __restrict__ oz, float* __restrict__ om,
                             float* __restrict__ os) {
    const int i = blockIdx.x * blockDim.x + threadIdx.x;
    if (i >= B_ * Z_) return;
    const int b = i / Z_, z = i % Z_;
    const float mu = musig[(size_t)b * 512 + z];
    const float sg = musig[(size_t)b * 512 + Z_ + z];
    const float zn = mu + expf(sg) * eps_i[i];
    zi[i] = zn;
    const bool valid = step < kv[b];
    const size_t o = (size_t)b * KMAX_ * Z_ + (size_t)step * Z_ + z;
    oz[o] = valid ? zn : 0.0f;
    om[o] = valid ? mu : 0.0f;
    os[o] = valid ? sg : 0.0f;
}

// ---------------------------------------------------------------------------
// One-time prep kernels: pack weights as bf16 [N,Kp] row-major, combine biases
// ---------------------------------------------------------------------------
__global__ void build_wg0_kernel(const float* __restrict__ wih, const float* __restrict__ whh,
                                 __bf16* __restrict__ dst) {
    const int i = blockIdx.x * blockDim.x + threadIdx.x;
    if (i >= G4H_ * KP0_) return;
    const int n = i / KP0_, kk = i % KP0_;
    float v = 0.0f;
    if (kk < IN_)            v = wih[(size_t)n * IN_ + kk];
    else if (kk < IN_ + H_)  v = whh[(size_t)n * H_ + (kk - IN_)];
    dst[i] = (__bf16)v;
}

__global__ void build_wg1_kernel(const float* __restrict__ wih, const float* __restrict__ whh,
                                 __bf16* __restrict__ dst) {
    const int i = blockIdx.x * blockDim.x + threadIdx.x;
    if (i >= G4H_ * KP1_) return;
    const int n = i / KP1_, kk = i % KP1_;
    const float v = (kk < H_) ? wih[(size_t)n * H_ + kk]
                              : whh[(size_t)n * H_ + (kk - H_)];
    dst[i] = (__bf16)v;
}

__global__ void build_wms_kernel(const float* __restrict__ muW, const float* __restrict__ sgW,
                                 __bf16* __restrict__ dst) {
    const int i = blockIdx.x * blockDim.x + threadIdx.x;
    if (i >= 512 * H_) return;
    const int n = i / H_, kk = i % H_;
    const float v = (n < Z_) ? muW[(size_t)n * H_ + kk]
                             : sgW[(size_t)(n - Z_) * H_ + kk];
    dst[i] = (__bf16)v;
}

__global__ void convert_bf16_kernel(const float* __restrict__ src, __bf16* __restrict__ dst, int n) {
    const int i = blockIdx.x * blockDim.x + threadIdx.x;
    if (i < n) dst[i] = (__bf16)src[i];
}

__global__ void build_bias_kernel(const float* bih0, const float* bhh0, float* b0,
                                  const float* bih1, const float* bhh1, float* b1,
                                  const float* mub, const float* sgb, float* bm) {
    const int i = blockIdx.x * blockDim.x + threadIdx.x;
    if (i < G4H_) { b0[i] = bih0[i] + bhh0[i]; b1[i] = bih1[i] + bhh1[i]; }
    if (i < 512)  bm[i] = (i < Z_) ? mub[i] : sgb[i - Z_];
}

__global__ void init_state_kernel(const float* h0, const float* c0, const float* z0,
                                  float* h0s, float* h1s, float* c0s, float* c1s,
                                  float* zi) {
    const int i = blockIdx.x * blockDim.x + threadIdx.x;
    if (i < B_ * H_) {
        h0s[i] = h0[i];              h1s[i] = h0[B_ * H_ + i];
        c0s[i] = c0[i];              c1s[i] = c0[B_ * H_ + i];
    }
    if (i < B_ * Z_) zi[i] = z0[i];
}

__global__ void finalize_kernel(const float* h0s, const float* h1s,
                                const float* c0s, const float* c1s,
                                float* __restrict__ o) {
    const int i = blockIdx.x * blockDim.x + threadIdx.x;
    if (i >= B_ * H_) return;
    o[i]               = h0s[i];
    o[B_ * H_ + i]     = h1s[i];
    o[2 * B_ * H_ + i] = c0s[i];
    o[3 * B_ * H_ + i] = c1s[i];
}

// ---------------------------------------------------------------------------
extern "C" void kernel_launch(void* const* d_in, const int* in_sizes, int n_in,
                              void* d_out, int out_size, void* d_ws, size_t ws_size,
                              hipStream_t stream) {
    const float* h0   = (const float*)d_in[0];
    const float* c0   = (const float*)d_in[1];
    const float* ctx  = (const float*)d_in[2];
    const float* z0   = (const float*)d_in[3];
    const int*   kv   = (const int*)  d_in[4];
    const float* eps  = (const float*)d_in[5];
    const float* Wih0 = (const float*)d_in[6];
    const float* Whh0 = (const float*)d_in[7];
    const float* bih0 = (const float*)d_in[8];
    const float* bhh0 = (const float*)d_in[9];
    const float* Wih1 = (const float*)d_in[10];
    const float* Whh1 = (const float*)d_in[11];
    const float* bih1 = (const float*)d_in[12];
    const float* bhh1 = (const float*)d_in[13];
    const float* Wina = (const float*)d_in[14];
    const float* Wout = (const float*)d_in[15];
    const float* muW  = (const float*)d_in[16];
    const float* mub  = (const float*)d_in[17];
    const float* sgW  = (const float*)d_in[18];
    const float* sgb  = (const float*)d_in[19];
    float* out = (float*)d_out;

    size_t off = 0;
    auto carve = [&](size_t bytes) -> void* {
        void* p = (char*)d_ws + off;
        off += (bytes + 255) & ~(size_t)255;
        return p;
    };

    __bf16* wg0  = (__bf16*)carve((size_t)G4H_ * KP0_ * 2);
    __bf16* wg1  = (__bf16*)carve((size_t)G4H_ * KP1_ * 2);
    __bf16* wms  = (__bf16*)carve((size_t)512 * H_ * 2);
    __bf16* win  = (__bf16*)carve((size_t)H_ * H_ * 2);
    __bf16* wou  = (__bf16*)carve((size_t)H_ * 2 * H_ * 2);
    float*  bias0 = (float*)carve(G4H_ * 4);
    float*  bias1 = (float*)carve(G4H_ * 4);
    float*  biasm = (float*)carve(512 * 4);
    __bf16* X0 = (__bf16*)carve((size_t)B_ * KP0_ * 2);
    __bf16* X1 = (__bf16*)carve((size_t)B_ * KP1_ * 2);
    __bf16* Xc = (__bf16*)carve((size_t)B_ * 2 * H_ * 2);   // [weighted | q]
    float* h0s  = (float*)carve((size_t)B_ * H_ * 4);
    float* h1s  = (float*)carve((size_t)B_ * H_ * 4);
    float* c0s  = (float*)carve((size_t)B_ * H_ * 4);
    float* c1s  = (float*)carve((size_t)B_ * H_ * 4);
    float* h0r  = (float*)carve((size_t)B_ * H_ * 4);
    float* h1r  = (float*)carve((size_t)B_ * H_ * 4);
    float* attn = (float*)carve((size_t)B_ * H_ * 4);
    float* qin  = (float*)carve((size_t)B_ * H_ * 4);
    float* zi   = (float*)carve((size_t)B_ * Z_ * 4);
    float* musig = (float*)carve((size_t)B_ * 512 * 4);

    const int T = 256;
    const size_t OZ = (size_t)B_ * KMAX_ * Z_;

    // One-time prep: bf16 weight packs, combined biases, initial state.
    build_wg0_kernel<<<CEILDIV(G4H_ * KP0_, T), T, 0, stream>>>(Wih0, Whh0, wg0);
    build_wg1_kernel<<<CEILDIV(G4H_ * KP1_, T), T, 0, stream>>>(Wih1, Whh1, wg1);
    build_wms_kernel<<<CEILDIV(512 * H_, T), T, 0, stream>>>(muW, sgW, wms);
    convert_bf16_kernel<<<CEILDIV(H_ * H_, T), T, 0, stream>>>(Wina, win, H_ * H_);
    convert_bf16_kernel<<<CEILDIV(H_ * 2 * H_, T), T, 0, stream>>>(Wout, wou, H_ * 2 * H_);
    build_bias_kernel<<<CEILDIV(G4H_, T), T, 0, stream>>>(bih0, bhh0, bias0,
                                                          bih1, bhh1, bias1,
                                                          mub, sgb, biasm);
    init_state_kernel<<<CEILDIV(B_ * H_, T), T, 0, stream>>>(h0, c0, z0, h0s, h1s, c0s, c1s, zi);

    // attn0 = attend(c0[last layer])
    pack_xcat_hi_kernel<<<CEILDIV(B_ * H_, T), T, 0, stream>>>(c0 + (size_t)B_ * H_, Xc);
    wmma_gemm_kernel<0><<<dim3(H_ / 16, B_ / 16), 32, 0, stream>>>(Xc + H_, 2 * H_, win, H_,
                                                                   nullptr, qin, H_);
    attn_scores_kernel<<<B_, 256, 0, stream>>>(ctx, qin, Xc);
    wmma_gemm_kernel<1><<<dim3(H_ / 16, B_ / 16), 32, 0, stream>>>(Xc, 2 * H_, wou, 2 * H_,
                                                                   nullptr, attn, H_);

    for (int step = 0; step < KMAX_; ++step) {
        pack_x0_kernel<<<CEILDIV(B_ * KP0_, T), T, 0, stream>>>(zi, kv, attn, h0s, X0);
        wmma_lstm_kernel<<<dim3(H_ / 16, B_ / 16), 32, 0, stream>>>(X0, KP0_, wg0, bias0,
                                                                    kv, step, h0s, c0s, h0r);
        pack_x1_kernel<<<CEILDIV(B_ * KP1_, T), T, 0, stream>>>(h0r, h1s, X1);
        wmma_lstm_kernel<<<dim3(H_ / 16, B_ / 16), 32, 0, stream>>>(X1, KP1_, wg1, bias1,
                                                                    kv, step, h1s, c1s, h1r);
        // output = raw h1; it feeds mu/sigma, q_in, and the concat for W_out
        pack_xcat_hi_kernel<<<CEILDIV(B_ * H_, T), T, 0, stream>>>(h1r, Xc);
        wmma_gemm_kernel<0><<<dim3(512 / 16, B_ / 16), 32, 0, stream>>>(Xc + H_, 2 * H_, wms, H_,
                                                                        biasm, musig, 512);
        zstep_kernel<<<CEILDIV(B_ * Z_, T), T, 0, stream>>>(musig, eps + (size_t)step * B_ * Z_,
                                                            kv, step, zi,
                                                            out, out + OZ, out + 2 * OZ);
        wmma_gemm_kernel<0><<<dim3(H_ / 16, B_ / 16), 32, 0, stream>>>(Xc + H_, 2 * H_, win, H_,
                                                                       nullptr, qin, H_);
        attn_scores_kernel<<<B_, 256, 0, stream>>>(ctx, qin, Xc);
        wmma_gemm_kernel<1><<<dim3(H_ / 16, B_ / 16), 32, 0, stream>>>(Xc, 2 * H_, wou, 2 * H_,
                                                                       nullptr, attn, H_);
    }

    finalize_kernel<<<CEILDIV(B_ * H_, T), T, 0, stream>>>(h0s, h1s, c0s, c1s, out + 3 * OZ);
}
